// LESSR_86440511799495
// MI455X (gfx1250) — matrix operand order repaired and checked by reading, not compiled, not run on hardware
//
#include <hip/hip_runtime.h>
#include <math.h>

typedef __attribute__((ext_vector_type(16))) __bf16 v16bf;
typedef __attribute__((ext_vector_type(8)))  float  v8f;

#define B_SZ 1024
#define NPG  20
#define NND  20480
#define DEG  4
#define NE   163840
#define DIM  128
#define NV   100000
#define NC   1000
#define NP   100
#define EPS  1e-5f

// ---------------- device helpers ----------------
__device__ __forceinline__ float sigm(float x){ return 1.0f/(1.0f+expf(-x)); }
__device__ __forceinline__ float wred(float v){
#pragma unroll
  for (int o = 16; o > 0; o >>= 1) v += __shfl_xor(v, o, 32);
  return v;
}
__device__ __forceinline__ unsigned fenc(float f){
  unsigned u = __float_as_uint(f);
  return (u & 0x80000000u) ? ~u : (u | 0x80000000u);
}
__device__ __forceinline__ float fdec(unsigned u){
  return __uint_as_float((u & 0x80000000u) ? (u ^ 0x80000000u) : ~u);
}

// ---------------- table renorm (max_norm=1) ----------------
__global__ void krow_scale(const float* __restrict__ tab, float* __restrict__ scale,
                           __bf16* __restrict__ outbf, int rows){
  int wid = threadIdx.x >> 5;
  int r = blockIdx.x * 8 + wid;
  if (r >= rows) return;
  int lane = threadIdx.x & 31;
  float x[4]; float s = 0.f;
#pragma unroll
  for (int j = 0; j < 4; j++){
    x[j] = tab[(size_t)r * DIM + lane + 32*j];
    s += x[j]*x[j];
  }
  s = wred(s);
  float nrm = sqrtf(s);
  float sc = fminf(1.0f, 1.0f / fmaxf(nrm, 1e-12f));
  if (outbf){
#pragma unroll
    for (int j = 0; j < 4; j++)
      outbf[(size_t)r * DIM + lane + 32*j] = (__bf16)(x[j]*sc);
  }
  if (lane == 0 && scale) scale[r] = sc;
}

__global__ void kgather_scaled(const float* __restrict__ tab, const float* __restrict__ scale,
                               const int* __restrict__ idx, float* __restrict__ out, int rows){
  int i = blockIdx.x * blockDim.x + threadIdx.x;
  if (i >= rows * DIM) return;
  int r = i >> 7, d = i & 127;
  int id = idx[r];
  out[i] = tab[(size_t)id * DIM + d] * scale[id];
}

// ---------------- weight transpose-convert: W[K,Nc] f32 -> Wt[Nc,K] bf16 ----------------
__global__ void kconv_t(const float* __restrict__ W, __bf16* __restrict__ Wt, int K, int Nc){
  int i = blockIdx.x * blockDim.x + threadIdx.x;
  if (i >= K * Nc) return;
  int k = i / Nc, n = i % Nc;
  Wt[(size_t)n * K + k] = (__bf16)W[i];
}

// ---------------- elementwise ----------------
__global__ void kfill(float* p, float v, int n){
  int i = blockIdx.x * blockDim.x + threadIdx.x; if (i < n) p[i] = v;
}
__global__ void kfillu(unsigned* p, unsigned v, int n){
  int i = blockIdx.x * blockDim.x + threadIdx.x; if (i < n) p[i] = v;
}
__global__ void kmul_inplace(float* x, const float* y, int n){
  int i = blockIdx.x * blockDim.x + threadIdx.x; if (i < n) x[i] *= y[i];
}
__global__ void krelu_copy(float* dst, const float* src, int n){
  int i = blockIdx.x * blockDim.x + threadIdx.x; if (i < n) dst[i] = fmaxf(src[i], 0.f);
}
__global__ void kconcat_copy(float* dst, int ldd, const float* src, int lds_,
                             int rows, int cols){
  int i = blockIdx.x * blockDim.x + threadIdx.x;
  if (i >= rows * cols) return;
  int r = i / cols, c = i % cols;
  dst[(size_t)r * ldd + c] = src[(size_t)r * lds_ + c];
}
__global__ void kprelu(float* x, int ld, int cols, const float* alpha, int rows){
  int i = blockIdx.x * blockDim.x + threadIdx.x;
  if (i >= rows * cols) return;
  int r = i / cols, c = i % cols;
  float v = x[(size_t)r * ld + c];
  x[(size_t)r * ld + c] = v > 0.f ? v : alpha[c] * v;
}

// ---------------- BatchNorm (train mode, biased var) ----------------
__global__ void kbn_stats(const float* __restrict__ X, int rows, int cols,
                          float* mean, float* var){
  __shared__ float sh[256], sh2[256];
  int c = blockIdx.x, tid = threadIdx.x;
  float s = 0.f, s2 = 0.f;
  for (int r = tid; r < rows; r += blockDim.x){
    float v = X[(size_t)r * cols + c];
    s += v; s2 += v*v;
  }
  sh[tid] = s; sh2[tid] = s2; __syncthreads();
  for (int o = 128; o > 0; o >>= 1){
    if (tid < o){ sh[tid] += sh[tid+o]; sh2[tid] += sh2[tid+o]; }
    __syncthreads();
  }
  if (tid == 0){
    float m = sh[0] / rows;
    mean[c] = m;
    var[c] = sh2[0] / rows - m*m;
  }
}
__global__ void kbn_apply(const float* __restrict__ X, float* __restrict__ Y,
                          int rows, int cols, const float* mean, const float* var,
                          const float* g, const float* b){
  int i = blockIdx.x * blockDim.x + threadIdx.x;
  if (i >= rows * cols) return;
  int c = i % cols;
  Y[i] = (X[i] - mean[c]) * rsqrtf(var[c] + EPS) * g[c] + b[c];
}
// Intend BN: per-timestep scalar over B x D, seq stored as catf [B,20,D]
__global__ void kbn_seq_stats(const float* __restrict__ seq, float* mean, float* var){
  __shared__ float sh[256], sh2[256];
  int t = blockIdx.x, tid = threadIdx.x;
  float s = 0.f, s2 = 0.f;
  for (int i = tid; i < B_SZ * DIM; i += blockDim.x){
    int b = i >> 7, d = i & 127;
    float v = seq[((size_t)(b * NPG + t)) * DIM + d];
    s += v; s2 += v*v;
  }
  sh[tid] = s; sh2[tid] = s2; __syncthreads();
  for (int o = 128; o > 0; o >>= 1){
    if (tid < o){ sh[tid] += sh[tid+o]; sh2[tid] += sh2[tid+o]; }
    __syncthreads();
  }
  if (tid == 0){
    float m = sh[0] / (B_SZ * DIM);
    mean[t] = m;
    var[t] = sh2[0] / (B_SZ * DIM) - m*m;
  }
}
__global__ void kbn_seq_apply(float* seq, const float* mean, const float* var,
                              const float* g, const float* b){
  int i = blockIdx.x * blockDim.x + threadIdx.x;
  if (i >= B_SZ * NPG * DIM) return;
  int t = (i / DIM) % NPG;
  seq[i] = (seq[i] - mean[t]) * rsqrtf(var[t] + EPS) * g[t] + b[t];
}

// ---------------- unified NT WMMA GEMM ----------------
// C[M,Nc] (+=) act( A[M,K]f32 @ Bt[Nc,K]bf16 ^T + bias ).  Optional A row gather.
// Per-lane fragment loads are fully contiguous:
//   A: two 32B f32 chunks at K = kk+16g and kk+8+16g  (halves: h+16g)
//   B: one 32B bf16 run at K = kk+16g               (halves: h+16g, N = lane&15)
__global__ void kgemm_nt(const float* __restrict__ A, int lda,
                         const int* __restrict__ aidx, int aidx_stride,
                         const __bf16* __restrict__ Bt,
                         const float* __restrict__ bias,
                         float* __restrict__ C, int ldc,
                         int M, int Nc, int K, int act, int accum){
  int wid = threadIdx.x >> 5;
  int tile = blockIdx.x * (blockDim.x >> 5) + wid;
  int tn = Nc >> 4;
  int total = (M >> 4) * tn;
  if (tile >= total) return;                 // wave-uniform: EXEC stays all-ones
  int m0 = (tile / tn) << 4;
  int n0 = (tile % tn) << 4;
  int lane = threadIdx.x & 31;
  int g = lane >> 4, l = lane & 15;
  int row = m0 + l;
  int rsel = aidx ? aidx[(size_t)row * aidx_stride] : row;
  const float*  arow = A  + (size_t)rsel * lda;
  const __bf16* brow = Bt + (size_t)(n0 + l) * K;
  __builtin_prefetch(arow, 0, 1);            // -> global_prefetch_b8
  __builtin_prefetch(brow, 0, 1);
  v8f acc = {};
  for (int kk = 0; kk < K; kk += 32){
    float ta[16];
    const float4* pa0 = (const float4*)(arow + kk + 16*g);
    ((float4*)ta)[0] = pa0[0];
    ((float4*)ta)[1] = pa0[1];
    const float4* pa1 = (const float4*)(arow + kk + 8 + 16*g);
    ((float4*)ta)[2] = pa1[0];
    ((float4*)ta)[3] = pa1[1];
    v16bf a;
#pragma unroll
    for (int h = 0; h < 16; h++) a[h] = (__bf16)ta[h];
    v16bf b = *(const v16bf*)(brow + kk + 16*g);
    acc = __builtin_amdgcn_wmma_f32_16x16x32_bf16(false, a, false, b, (short)0, acc, false, false);
  }
  float bv = bias ? bias[n0 + l] : 0.f;
#pragma unroll
  for (int i = 0; i < 8; i++){
    size_t idx = (size_t)(m0 + i + 8*g) * ldc + n0 + l;  // C: M = i+8g, N = l
    float r = acc[i] + bv;
    if (accum) r += C[idx];
    if (act == 1) r = sigm(r);
    C[idx] = r;
  }
}

// ---------------- GRU gate combine (torch order r,z,n) ----------------
__global__ void kgru_gate(const float* __restrict__ xw, const float* __restrict__ hw,
                          float* __restrict__ h, int rows, int H){
  int i = blockIdx.x * blockDim.x + threadIdx.x;
  if (i >= rows * H) return;
  int r = i / H, j = i % H;
  const float* xp = xw + (size_t)r * 3 * H;
  const float* hp = hw + (size_t)r * 3 * H;
  float rg = sigm(xp[j] + hp[j]);
  float zg = sigm(xp[H + j] + hp[H + j]);
  float ng = tanhf(xp[2*H + j] + rg * hp[2*H + j]);
  float hv = h[(size_t)r * H + j];
  h[(size_t)r * H + j] = (1.f - zg) * ng + zg * hv;
}

// ---------------- edge attention / segment softmax ----------------
__global__ void kedge_e(const float* __restrict__ q, const float* __restrict__ k,
                        const float* __restrict__ We, const int* __restrict__ src,
                        const int* __restrict__ dst, float* __restrict__ e, int nE){
  int wid = threadIdx.x >> 5;
  int eid = blockIdx.x * 8 + wid;
  if (eid >= nE) return;
  int lane = threadIdx.x & 31;
  int s = src[eid], d = dst[eid];
  float acc = 0.f;
#pragma unroll
  for (int j = 0; j < 4; j++){
    int dd = lane + 32*j;
    acc += We[dd] * sigm(q[(size_t)s * DIM + dd] + k[(size_t)d * DIM + dd]);
  }
  acc = wred(acc);
  if (lane == 0) e[eid] = acc;
}
__global__ void kedge_max(const float* __restrict__ e, const int* __restrict__ dst,
                          unsigned* __restrict__ menc, int nE){
  int i = blockIdx.x * blockDim.x + threadIdx.x;
  if (i < nE) atomicMax(&menc[dst[i]], fenc(e[i]));
}
__global__ void kedge_exp(float* __restrict__ e, const int* __restrict__ dst,
                          const unsigned* __restrict__ menc, float* __restrict__ s, int nE){
  int i = blockIdx.x * blockDim.x + threadIdx.x;
  if (i >= nE) return;
  float ex = expf(e[i] - fdec(menc[dst[i]]));
  e[i] = ex;
  atomicAdd(&s[dst[i]], ex);
}
__global__ void kedge_scatter(const float* __restrict__ ex, const float* __restrict__ s,
                              const int* __restrict__ src, const int* __restrict__ dst,
                              const float* __restrict__ vv, float* __restrict__ rst, int nE){
  int wid = threadIdx.x >> 5;
  int eid = blockIdx.x * 8 + wid;
  if (eid >= nE) return;
  int lane = threadIdx.x & 31;
  int sr = src[eid], d = dst[eid];
  float a = ex[eid] / fmaxf(s[d], 1e-12f);
#pragma unroll
  for (int j = 0; j < 4; j++){
    int dd = lane + 32*j;
    atomicAdd(&rst[(size_t)d * DIM + dd], vv[(size_t)sr * DIM + dd] * a);
  }
}

// ---------------- readout ----------------
__global__ void ker_node(const float* __restrict__ fu, const float* __restrict__ fv,
                         const float* __restrict__ le, const float* __restrict__ Wer,
                         const int* __restrict__ seg, float* __restrict__ er, int n){
  int wid = threadIdx.x >> 5;
  int nd = blockIdx.x * 8 + wid;
  if (nd >= n) return;
  int lane = threadIdx.x & 31;
  int g = seg[nd];
  float acc = 0.f;
#pragma unroll
  for (int j = 0; j < 4; j++){
    int dd = lane + 32*j;
    acc += Wer[dd] * sigm(fu[(size_t)nd * DIM + dd] + fv[(size_t)g * DIM + dd]
                          + le[(size_t)g * DIM + dd]);
  }
  acc = wred(acc);
  if (lane == 0) er[nd] = acc;
}
__global__ void kreadout(const float* __restrict__ er, const float* __restrict__ fr,
                         float* __restrict__ srg){
  int b = blockIdx.x;
  const float* eb = er + (size_t)b * NPG;
  float m = -1e30f;
  for (int j = 0; j < NPG; j++) m = fmaxf(m, eb[j]);
  float al[NPG]; float ssum = 0.f;
  for (int j = 0; j < NPG; j++){ al[j] = expf(eb[j] - m); ssum += al[j]; }
  float inv = 1.f / fmaxf(ssum, 1e-12f);
  for (int c = threadIdx.x; c < 3*DIM; c += blockDim.x){
    float acc = 0.f;
    for (int j = 0; j < NPG; j++)
      acc += fr[((size_t)(b * NPG + j)) * (3*DIM) + c] * al[j];
    srg[(size_t)b * (3*DIM) + c] = acc * inv;
  }
}

// ---------------- host launcher ----------------
static inline int cdiv(long long a, long long b){ return (int)((a + b - 1) / b); }

extern "C" void kernel_launch(void* const* d_in, const int* in_sizes, int n_in,
                              void* d_out, int out_size, void* d_ws, size_t ws_size,
                              hipStream_t stream){
  const float* emb_item = (const float*)d_in[0];
  const float* emb_cat  = (const float*)d_in[1];
  const float* emb_price= (const float*)d_in[2];
  const float* bn_pa_g = (const float*)d_in[3];
  const float* bn_pa_b = (const float*)d_in[4];
  const float* pa_W    = (const float*)d_in[5];
  const float* pa_b    = (const float*)d_in[6];
  const float* bn_int_g= (const float*)d_in[7];
  const float* bn_int_b= (const float*)d_in[8];
  const float* int_Wi  = (const float*)d_in[9];
  const float* int_Wh  = (const float*)d_in[10];
  const float* int_bi  = (const float*)d_in[11];
  const float* int_bh  = (const float*)d_in[12];
  const float* bn0_g   = (const float*)d_in[13];
  const float* bn0_b   = (const float*)d_in[14];
  const float* gru0_Wi = (const float*)d_in[15];
  const float* gru0_Wh = (const float*)d_in[16];
  const float* gru0_bi = (const float*)d_in[17];
  const float* gru0_bh = (const float*)d_in[18];
  const float* W_self  = (const float*)d_in[19];
  const float* W_neigh = (const float*)d_in[20];
  const float* alpha0  = (const float*)d_in[21];
  const float* bn1_g   = (const float*)d_in[22];
  const float* bn1_b   = (const float*)d_in[23];
  const float* Wq      = (const float*)d_in[24];
  const float* bq      = (const float*)d_in[25];
  const float* Wk      = (const float*)d_in[26];
  const float* Wv      = (const float*)d_in[27];
  const float* We      = (const float*)d_in[28];
  const float* alpha1  = (const float*)d_in[29];
  const float* bn_ro_g = (const float*)d_in[30];
  const float* bn_ro_b = (const float*)d_in[31];
  const float* Wu      = (const float*)d_in[32];
  const float* Wv_r    = (const float*)d_in[33];
  const float* bv_r    = (const float*)d_in[34];
  const float* Wi_r    = (const float*)d_in[35];
  const float* bi_r    = (const float*)d_in[36];
  const float* We_r    = (const float*)d_in[37];
  const float* Wout    = (const float*)d_in[38];
  const float* alpha_ro= (const float*)d_in[39];
  const float* bn_sr_g = (const float*)d_in[40];
  const float* bn_sr_b = (const float*)d_in[41];
  const float* W_sr    = (const float*)d_in[42];
  const int* iid       = (const int*)d_in[43];
  const int* cid       = (const int*)d_in[44];
  const int* pid       = (const int*)d_in[45];
  const int* nbr_idx   = (const int*)d_in[47];
  const int* sg_src    = (const int*)d_in[48];
  const int* sg_dst    = (const int*)d_in[49];
  const int* seg_ids   = (const int*)d_in[50];
  const int* last_nodes= (const int*)d_in[51];
  float* out = (float*)d_out;

  // ---- workspace layout ----
  char* ws = (char*)d_ws;
  size_t off = 0;
  auto take = [&](size_t bytes) -> char* {
    char* p = ws + off;
    off += (bytes + 255) & ~(size_t)255;
    return p;
  };
  __bf16*   item_bf   = (__bf16*)take((size_t)NV * DIM * 2);
  float*    item_sc   = (float*) take((size_t)NV * 4);
  float*    cat_sc    = (float*) take((size_t)NC * 4);
  float*    price_sc  = (float*) take((size_t)NP * 4);
  float*    feat      = (float*) take((size_t)NND * DIM * 4);
  float*    catf      = (float*) take((size_t)NND * DIM * 4);   // -> seqn in place
  float*    pcn       = (float*) take((size_t)NND * DIM * 4);
  float*    pfac      = (float*) take((size_t)NND * 3*DIM * 4);
  float*    bigH      = (float*) take((size_t)NND * 3*DIM * 4); // xw / feat3
  float*    bigI      = (float*) take((size_t)NND * 3*DIM * 4); // hw / fr
  float*    hJ        = (float*) take((size_t)NND * DIM * 4);   // GRU hidden / neigh
  float*    intend    = (float*) take((size_t)B_SZ * 3*DIM * 4);
  float*    f0        = (float*) take((size_t)NND * DIM * 4);
  float*    feat2     = (float*) take((size_t)NND * 2*DIM * 4);
  float*    f1        = (float*) take((size_t)NND * 2*DIM * 4);
  float*    qb        = (float*) take((size_t)NND * DIM * 4);   // q / feat_u
  float*    kb        = (float*) take((size_t)NND * DIM * 4);
  float*    vb        = (float*) take((size_t)NND * DIM * 4);
  float*    rst       = (float*) take((size_t)NND * DIM * 4);
  float*    ebuf      = (float*) take((size_t)NE * 4);
  float*    sbuf      = (float*) take((size_t)NND * 4);
  unsigned* menc      = (unsigned*)take((size_t)NND * 4);
  float*    lastB     = (float*) take((size_t)B_SZ * DIM * 4);
  float*    fvB       = (float*) take((size_t)B_SZ * DIM * 4);
  float*    erN       = (float*) take((size_t)NND * 4);
  float*    srgB      = (float*) take((size_t)B_SZ * 3*DIM * 4);
  float*    srg2B     = (float*) take((size_t)B_SZ * DIM * 4);
  float*    srB       = (float*) take((size_t)B_SZ * 4*DIM * 4);
  float*    srwB      = (float*) take((size_t)B_SZ * DIM * 4);
  float*    meanb     = (float*) take(1152 * 4);
  float*    varb      = (float*) take(1152 * 4);
  // bf16 transposed weights [Nc, K]
  __bf16* paT    = (__bf16*)take((size_t)3*DIM * DIM   * 2);
  __bf16* intWiT = (__bf16*)take((size_t)9*DIM * DIM   * 2);
  __bf16* intWhT = (__bf16*)take((size_t)9*DIM * 3*DIM * 2);
  __bf16* g0WiT  = (__bf16*)take((size_t)3*DIM * DIM   * 2);
  __bf16* g0WhT  = (__bf16*)take((size_t)3*DIM * DIM   * 2);
  __bf16* WselfT = (__bf16*)take((size_t)DIM   * DIM   * 2);
  __bf16* WneighT= (__bf16*)take((size_t)DIM   * DIM   * 2);
  __bf16* WqT    = (__bf16*)take((size_t)DIM   * 2*DIM * 2);
  __bf16* WkT    = (__bf16*)take((size_t)DIM   * 2*DIM * 2);
  __bf16* WvT    = (__bf16*)take((size_t)DIM   * 2*DIM * 2);
  __bf16* WuT    = (__bf16*)take((size_t)DIM   * 3*DIM * 2);
  __bf16* WvrT   = (__bf16*)take((size_t)DIM   * 3*DIM * 2);
  __bf16* WirT   = (__bf16*)take((size_t)DIM   * 3*DIM * 2);
  __bf16* WoutT  = (__bf16*)take((size_t)DIM   * 3*DIM * 2);
  __bf16* WsrT   = (__bf16*)take((size_t)DIM   * 4*DIM * 2);
  (void)ws_size; (void)n_in; (void)in_sizes; (void)out_size;

  auto gemm = [&](const float* A, int lda, const int* aidx, int astr,
                  const __bf16* Bt, const float* bias, float* C, int ldc,
                  int M, int Nc, int K, int act, int accum){
    long long tiles = (long long)(M/16) * (Nc/16);
    kgemm_nt<<<cdiv(tiles, 8), 256, 0, stream>>>(A, lda, aidx, astr, Bt, bias, C, ldc,
                                                 M, Nc, K, act, accum);
  };
  auto ew = [&](long long n){ return cdiv(n, 256); };
  auto conv = [&](const float* W, __bf16* Wt, int K, int Nc){
    kconv_t<<<ew((long long)K*Nc),256,0,stream>>>(W, Wt, K, Nc);
  };

  // ---- one-time weight transposes to bf16 [Nc,K] ----
  conv(pa_W,    paT,    DIM,   3*DIM);
  conv(int_Wi,  intWiT, DIM,   9*DIM);
  conv(int_Wh,  intWhT, 3*DIM, 9*DIM);
  conv(gru0_Wi, g0WiT,  DIM,   3*DIM);
  conv(gru0_Wh, g0WhT,  DIM,   3*DIM);
  conv(W_self,  WselfT, DIM,   DIM);
  conv(W_neigh, WneighT,DIM,   DIM);
  conv(Wq,      WqT,    2*DIM, DIM);
  conv(Wk,      WkT,    2*DIM, DIM);
  conv(Wv,      WvT,    2*DIM, DIM);
  conv(Wu,      WuT,    3*DIM, DIM);
  conv(Wv_r,    WvrT,   3*DIM, DIM);
  conv(Wi_r,    WirT,   3*DIM, DIM);
  conv(Wout,    WoutT,  3*DIM, DIM);
  conv(W_sr,    WsrT,   4*DIM, DIM);

  // ---- embedding renorm + gathers ----
  krow_scale<<<cdiv(NV,8),256,0,stream>>>(emb_item, item_sc, item_bf, NV);
  krow_scale<<<cdiv(NC,8),256,0,stream>>>(emb_cat,  cat_sc,  nullptr, NC);
  krow_scale<<<cdiv(NP,8),256,0,stream>>>(emb_price,price_sc,nullptr, NP);
  kgather_scaled<<<ew((long long)NND*DIM),256,0,stream>>>(emb_item, item_sc, iid, feat, NND);
  kgather_scaled<<<ew((long long)NND*DIM),256,0,stream>>>(emb_cat,  cat_sc,  cid, catf, NND);
  kgather_scaled<<<ew((long long)NND*DIM),256,0,stream>>>(emb_price,price_sc,pid, pcn,  NND);

  // ---- PriceAware: bn(price*cat) -> sigmoid(linear) ----
  kmul_inplace<<<ew((long long)NND*DIM),256,0,stream>>>(pcn, catf, NND*DIM);
  kbn_stats<<<DIM,256,0,stream>>>(pcn, NND, DIM, meanb, varb);
  kbn_apply<<<ew((long long)NND*DIM),256,0,stream>>>(pcn, pcn, NND, DIM, meanb, varb, bn_pa_g, bn_pa_b);
  gemm(pcn, DIM, nullptr, 0, paT, pa_b, pfac, 3*DIM, NND, 3*DIM, DIM, 1, 0);

  // ---- Intend: seq BN (per-t over BxD) + GRU(D->3D), relu ----
  kbn_seq_stats<<<NPG,256,0,stream>>>(catf, meanb, varb);
  kbn_seq_apply<<<ew((long long)B_SZ*NPG*DIM),256,0,stream>>>(catf, meanb, varb, bn_int_g, bn_int_b);
  kfill<<<ew((long long)B_SZ*3*DIM),256,0,stream>>>(hJ, 0.f, B_SZ*3*DIM);
  for (int t = 0; t < NPG; t++){
    gemm(catf + t*DIM, NPG*DIM, nullptr, 0, intWiT, int_bi, bigH, 9*DIM, B_SZ, 9*DIM, DIM,   0, 0);
    gemm(hJ,           3*DIM,   nullptr, 0, intWhT, int_bh, bigI, 9*DIM, B_SZ, 9*DIM, 3*DIM, 0, 0);
    kgru_gate<<<ew((long long)B_SZ*3*DIM),256,0,stream>>>(bigH, bigI, hJ, B_SZ, 3*DIM);
  }
  krelu_copy<<<ew((long long)B_SZ*3*DIM),256,0,stream>>>(intend, hJ, B_SZ*3*DIM);

  // ---- Layer 0: EOPA ----
  kbn_stats<<<DIM,256,0,stream>>>(feat, NND, DIM, meanb, varb);
  kbn_apply<<<ew((long long)NND*DIM),256,0,stream>>>(feat, f0, NND, DIM, meanb, varb, bn0_g, bn0_b);
  kfill<<<ew((long long)NND*DIM),256,0,stream>>>(hJ, 0.f, NND*DIM);
  for (int t = 0; t < DEG; t++){
    gemm(f0, DIM, nbr_idx + t, DEG, g0WiT, gru0_bi, bigH, 3*DIM, NND, 3*DIM, DIM, 0, 0);
    gemm(hJ, DIM, nullptr, 0,       g0WhT, gru0_bh, bigI, 3*DIM, NND, 3*DIM, DIM, 0, 0);
    kgru_gate<<<ew((long long)NND*DIM),256,0,stream>>>(bigH, bigI, hJ, NND, DIM);
  }
  gemm(f0, DIM, nullptr, 0, WselfT,  nullptr, feat2, 2*DIM, NND, DIM, DIM, 0, 0);
  gemm(hJ, DIM, nullptr, 0, WneighT, nullptr, feat2, 2*DIM, NND, DIM, DIM, 0, 1);
  kprelu<<<ew((long long)NND*DIM),256,0,stream>>>(feat2, 2*DIM, DIM, alpha0, NND);
  kconcat_copy<<<ew((long long)NND*DIM),256,0,stream>>>(feat2 + DIM, 2*DIM, feat, DIM, NND, DIM);

  // ---- Layer 1: SGAT ----
  kbn_stats<<<2*DIM,256,0,stream>>>(feat2, NND, 2*DIM, meanb, varb);
  kbn_apply<<<ew((long long)NND*2*DIM),256,0,stream>>>(feat2, f1, NND, 2*DIM, meanb, varb, bn1_g, bn1_b);
  gemm(f1, 2*DIM, nullptr, 0, WqT, bq,      qb, DIM, NND, DIM, 2*DIM, 0, 0);
  gemm(f1, 2*DIM, nullptr, 0, WkT, nullptr, kb, DIM, NND, DIM, 2*DIM, 0, 0);
  gemm(f1, 2*DIM, nullptr, 0, WvT, nullptr, vb, DIM, NND, DIM, 2*DIM, 0, 0);
  kedge_e<<<cdiv(NE,8),256,0,stream>>>(qb, kb, We, sg_src, sg_dst, ebuf, NE);
  kfillu<<<ew(NND),256,0,stream>>>(menc, 0x007FFFFFu, NND);   // encode(-inf)
  kedge_max<<<ew(NE),256,0,stream>>>(ebuf, sg_dst, menc, NE);
  kfill<<<ew(NND),256,0,stream>>>(sbuf, 0.f, NND);
  kedge_exp<<<ew(NE),256,0,stream>>>(ebuf, sg_dst, menc, sbuf, NE);
  kfill<<<ew((long long)NND*DIM),256,0,stream>>>(rst, 0.f, NND*DIM);
  kedge_scatter<<<cdiv(NE,8),256,0,stream>>>(ebuf, sbuf, sg_src, sg_dst, vb, rst, NE);
  kprelu<<<ew((long long)NND*DIM),256,0,stream>>>(rst, DIM, DIM, alpha1, NND);
  // feat3 = concat(out1, feat2) * price_factor  (bigH)
  kconcat_copy<<<ew((long long)NND*DIM),256,0,stream>>>(bigH, 3*DIM, rst, DIM, NND, DIM);
  kconcat_copy<<<ew((long long)NND*2*DIM),256,0,stream>>>(bigH + DIM, 3*DIM, feat2, 2*DIM, NND, 2*DIM);
  kmul_inplace<<<ew((long long)NND*3*DIM),256,0,stream>>>(bigH, pfac, NND*3*DIM);

  // ---- AttnReadout ----
  kbn_stats<<<3*DIM,256,0,stream>>>(bigH, NND, 3*DIM, meanb, varb);
  kbn_apply<<<ew((long long)NND*3*DIM),256,0,stream>>>(bigH, bigI, NND, 3*DIM, meanb, varb, bn_ro_g, bn_ro_b);
  gemm(bigI, 3*DIM, nullptr, 0,      WuT,  nullptr, qb,    DIM, NND,  DIM, 3*DIM, 0, 0);
  gemm(bigI, 3*DIM, last_nodes, 1,   WirT, bi_r,    lastB, DIM, B_SZ, DIM, 3*DIM, 0, 0);
  gemm(intend, 3*DIM, nullptr, 0,    WvrT, bv_r,    fvB,   DIM, B_SZ, DIM, 3*DIM, 0, 0);
  ker_node<<<cdiv(NND,8),256,0,stream>>>(qb, fvB, lastB, We_r, seg_ids, erN, NND);
  kreadout<<<B_SZ,128,0,stream>>>(erN, bigI, srgB);
  gemm(srgB, 3*DIM, nullptr, 0, WoutT, nullptr, srg2B, DIM, B_SZ, DIM, 3*DIM, 0, 0);
  kprelu<<<ew((long long)B_SZ*DIM),256,0,stream>>>(srg2B, DIM, DIM, alpha_ro, B_SZ);

  // ---- sr = bn(concat(intend, sr_g)); srW = sr @ W_sr ----
  kconcat_copy<<<ew((long long)B_SZ*3*DIM),256,0,stream>>>(srB, 4*DIM, intend, 3*DIM, B_SZ, 3*DIM);
  kconcat_copy<<<ew((long long)B_SZ*DIM),256,0,stream>>>(srB + 3*DIM, 4*DIM, srg2B, DIM, B_SZ, DIM);
  kbn_stats<<<4*DIM,256,0,stream>>>(srB, B_SZ, 4*DIM, meanb, varb);
  kbn_apply<<<ew((long long)B_SZ*4*DIM),256,0,stream>>>(srB, srB, B_SZ, 4*DIM, meanb, varb, bn_sr_g, bn_sr_b);
  gemm(srB, 4*DIM, nullptr, 0, WsrT, nullptr, srwB, DIM, B_SZ, DIM, 4*DIM, 0, 0);

  // ---- logits = srW @ item_tab^T : same NT kernel, Bt = renormed bf16 item table ----
  gemm(srwB, DIM, nullptr, 0, item_bf, nullptr, out, NV, B_SZ, NV, DIM, 0, 0);
}
